// MultiHeadDuplexAttention_1219770712284
// MI455X (gfx1250) — matrix operand order repaired and checked by reading, not compiled, not run on hardware
//
#include <hip/hip_runtime.h>
#include <math.h>

// ---------- types ----------
typedef __attribute__((ext_vector_type(16))) __bf16 bf16x16;
typedef __attribute__((ext_vector_type(8)))  __bf16 bf16x8;
typedef __attribute__((ext_vector_type(4)))  __bf16 bf16x4;
typedef __attribute__((ext_vector_type(8)))  float  f32x8;

#define D_MODEL 1024
#define NUM_HEADS 16
#define D_K 64
#define BATCH 8
#define SEQ 1024
#define M_TOK (BATCH * SEQ)           // 8192 rows
#define SQRT_DK 8.0f
#define INV_SQRT_DK 0.125f

__device__ __forceinline__ __bf16 f2bf(float f) { return (__bf16)f; }

// A-fragment (16-bit, 16x32): lane<16 holds K = 0..15 contiguous, lane>=16 holds K = 16..31 (ak = 0 or 16)
__device__ __forceinline__ bf16x16 load_a_frag(const __bf16* p) {
    bf16x8 lo = *(const bf16x8*)(p);
    bf16x8 hi = *(const bf16x8*)(p + 8);
    bf16x16 r;
#pragma unroll
    for (int i = 0; i < 8; ++i) { r[i] = lo[i]; r[i + 8] = hi[i]; }
    return r;
}

// B-fragment (16-bit, 32x16): lane<16 holds K=0..15 contiguous, lane>=16 holds K=16..31
__device__ __forceinline__ bf16x16 load_b_frag(const __bf16* p) {
    bf16x8 lo = *(const bf16x8*)(p);
    bf16x8 hi = *(const bf16x8*)(p + 8);
    bf16x16 r;
#pragma unroll
    for (int i = 0; i < 8; ++i) { r[i] = lo[i]; r[i + 8] = hi[i]; }
    return r;
}

__device__ __forceinline__ f32x8 wmma_bf16(bf16x16 a, bf16x16 b, f32x8 c) {
    return __builtin_amdgcn_wmma_f32_16x16x32_bf16(false, a, false, b, (short)0, c, false, false);
}

// ---------- prep kernels ----------
// Transpose fp32 weights [K][N] -> bf16 [N][K]; fold gamma/beta.
__global__ void prep_weights(const float* __restrict__ Wq, const float* __restrict__ Wk,
                             const float* __restrict__ Wv, const float* __restrict__ Wg,
                             const float* __restrict__ Wb, const float* __restrict__ Wo,
                             const float* __restrict__ bg, const float* __restrict__ bb,
                             __bf16* wqt, __bf16* wkt, __bf16* wvt, __bf16* wgbt, __bf16* wot,
                             float* bgb) {
    int idx = blockIdx.x * blockDim.x + threadIdx.x;       // 0 .. 1024*1024-1
    int n = idx & (D_MODEL - 1);
    int k = idx >> 10;
    int src = k * D_MODEL + n;
    int dst = n * D_MODEL + k;
    wqt[dst]  = f2bf(Wq[src]);
    wkt[dst]  = f2bf(Wk[src]);
    wvt[dst]  = f2bf(Wv[src]);
    wgbt[dst] = f2bf(SQRT_DK * Wg[src] + Wb[src]);
    wot[dst]  = f2bf(Wo[src]);
    if (idx < D_MODEL) bgb[idx] = SQRT_DK * bg[idx] + bb[idx];
}

__global__ void cvt_f32_bf16(const float* __restrict__ in, __bf16* __restrict__ out) {
    int i = blockIdx.x * blockDim.x + threadIdx.x;         // one float4 per thread
    float4 v = ((const float4*)in)[i];
    bf16x4 o; o[0] = f2bf(v.x); o[1] = f2bf(v.y); o[2] = f2bf(v.z); o[3] = f2bf(v.w);
    ((bf16x4*)out)[i] = o;
}

// ---------- generic WMMA GEMM:  C[8192x1024] = A[8192x1024] * W + bias ----------
// Wt is pre-transposed [N][K] bf16.  B tiles (64 cols x 128 K) are streamed into LDS
// with CDNA5 async loads (ASYNCcnt), double-buffered, and shared by all 8 waves.
// Store modes:
//   0: bf16 row-major [M][1024]
//   1: bf16 per-head Q/K layout  q[((b*16+h)*1024+s)*64+d]
//   2: bf16 transposed-V layout  v[((b*16+h)*64+d)*1024+s]
//   3: fp32 row-major
//   4: fp32 row-major + bf16 row-major copy
__global__ __launch_bounds__(256)
void gemm_bf16(const __bf16* __restrict__ A, const __bf16* __restrict__ Wt,
               const float* __restrict__ bias, __bf16* __restrict__ obf,
               float* __restrict__ ofp, int mode) {
    __shared__ __align__(16) __bf16 Bt[2][64][128];   // 2 x 16KB double buffer

    const int t    = threadIdx.x;
    const int lane = t & 31;
    const int wv   = t >> 5;
    const int l16  = lane & 15;
    const int hi   = lane >> 4;            // 0 or 1
    const int ak   = hi ? 16 : 0;
    const int kb   = hi ? 16 : 0;

    const int R  = blockIdx.x * 128 + wv * 16;   // 16-row strip for this wave
    const int Cb = blockIdx.y * 64;              // 64-col strip (shared by block)

    // async staging assignment: thread -> (col, 64B segment)
    const int lcol = t >> 2;                      // 0..63
    const int lseg = t & 3;                       // 0..3
    const __bf16* gsrc = Wt + (size_t)(Cb + lcol) * D_MODEL + lseg * 32;

    auto issue_b = [&](int buf, int kofs) {
        unsigned lds = (unsigned)(uintptr_t)&Bt[buf][lcol][lseg * 32];
        unsigned long long ga = (unsigned long long)(uintptr_t)(gsrc + kofs);
        asm volatile(
            "global_load_async_to_lds_b128 %0, %1, off\n\t"
            "global_load_async_to_lds_b128 %0, %1, off offset:16\n\t"
            "global_load_async_to_lds_b128 %0, %1, off offset:32\n\t"
            "global_load_async_to_lds_b128 %0, %1, off offset:48"
            :: "v"(lds), "v"(ga) : "memory");
    };

    const __bf16* Ar = A + (size_t)(R + l16) * D_MODEL;

    f32x8 acc[4] = {f32x8(0.f), f32x8(0.f), f32x8(0.f), f32x8(0.f)};

    issue_b(0, 0);
    asm volatile("s_wait_asynccnt 0x0" ::: "memory");
    __syncthreads();

    for (int ko = 0; ko < D_MODEL / 128; ++ko) {
        const int cur = ko & 1;
        if (ko < D_MODEL / 128 - 1) issue_b(cur ^ 1, (ko + 1) * 128);
#pragma unroll
        for (int ks = 0; ks < 4; ++ks) {
            const int k = ko * 128 + ks * 32;
            // batch all loads into distinct registers so ds_loads clause together
            // and waits become partial counts overlapping with WMMA execution
            bf16x16 a   = load_a_frag(Ar + k + ak);
            bf16x16 b0  = load_b_frag(&Bt[cur][0 * 16 + l16][ks * 32 + kb]);
            bf16x16 b1  = load_b_frag(&Bt[cur][1 * 16 + l16][ks * 32 + kb]);
            bf16x16 b2  = load_b_frag(&Bt[cur][2 * 16 + l16][ks * 32 + kb]);
            bf16x16 b3  = load_b_frag(&Bt[cur][3 * 16 + l16][ks * 32 + kb]);
            acc[0] = wmma_bf16(a, b0, acc[0]);
            acc[1] = wmma_bf16(a, b1, acc[1]);
            acc[2] = wmma_bf16(a, b2, acc[2]);
            acc[3] = wmma_bf16(a, b3, acc[3]);
        }
        asm volatile("s_wait_asynccnt 0x0" ::: "memory");
        __syncthreads();
    }

#pragma unroll
    for (int f = 0; f < 4; ++f) {
        const int col = Cb + f * 16 + l16;
        const float bval = bias[col];
#pragma unroll
        for (int r = 0; r < 8; ++r) {
            const int row = R + r + (hi ? 8 : 0);
            const float v = acc[f][r] + bval;
            if (mode == 0) {
                obf[(size_t)row * D_MODEL + col] = f2bf(v);
            } else if (mode == 1) {
                int b = row >> 10, s = row & (SEQ - 1), h = col >> 6, d = col & (D_K - 1);
                obf[(((size_t)(b * NUM_HEADS + h)) * SEQ + s) * D_K + d] = f2bf(v);
            } else if (mode == 2) {
                int b = row >> 10, s = row & (SEQ - 1), h = col >> 6, d = col & (D_K - 1);
                obf[(((size_t)(b * NUM_HEADS + h)) * D_K + d) * SEQ + s] = f2bf(v);
            } else if (mode == 3) {
                ofp[(size_t)row * D_MODEL + col] = v;
            } else {
                ofp[(size_t)row * D_MODEL + col] = v;
                obf[(size_t)row * D_MODEL + col] = f2bf(v);
            }
        }
    }
}

// ---------- flash attention:  one wave per (b,h, 16-query tile) ----------
// Q: [bh][s][d] bf16, K: [bh][s][d] bf16, Vt: [bh][d][s] bf16, ctx: [8192][1024] bf16
__global__ __launch_bounds__(256)
void attn_flash(const __bf16* __restrict__ Q, const __bf16* __restrict__ K,
                const __bf16* __restrict__ Vt, __bf16* __restrict__ ctx) {
    __shared__ __align__(16) __bf16 plds[8][16 * 32];   // per-wave P staging (C-layout -> A-layout)

    const int lane = threadIdx.x & 31;
    const int wv   = threadIdx.x >> 5;
    const int l16  = lane & 15;
    const int hi   = lane >> 4;
    const int ak   = hi ? 16 : 0;
    const int kb   = hi ? 16 : 0;

    const int tile  = blockIdx.x * 8 + wv;     // 0 .. 511
    const int bh    = tile >> 6;               // 0 .. 7
    const int qbase = (tile & 63) * 16;

    const __bf16* Qr = Q + ((size_t)bh * SEQ + qbase + l16) * D_K;
    const __bf16* Kb = K + (size_t)bh * SEQ * D_K;
    const __bf16* Vb = Vt + (size_t)bh * D_K * SEQ;

    const bf16x16 aq0 = load_a_frag(Qr + ak);        // d 0..31
    const bf16x16 aq1 = load_a_frag(Qr + 32 + ak);   // d 32..63

    f32x8 acc[4] = {f32x8(0.f), f32x8(0.f), f32x8(0.f), f32x8(0.f)};
    float m[8], l[8];
#pragma unroll
    for (int r = 0; r < 8; ++r) { m[r] = -3.0e38f; l[r] = 0.f; }

    __bf16* P = plds[wv];

    for (int kc = 0; kc < SEQ / 32; ++kc) {
        const int k0 = kc * 32;
        // ---- scores S = Q K^T / sqrt(dk): two 16x16 tiles (keys k0..+15, k0+16..+31)
        const __bf16* K0 = Kb + (size_t)(k0 + l16) * D_K;
        const __bf16* K1 = Kb + (size_t)(k0 + 16 + l16) * D_K;
        bf16x16 bk0 = load_b_frag(K0 + kb);
        bf16x16 bk1 = load_b_frag(K0 + 32 + kb);
        bf16x16 bk2 = load_b_frag(K1 + kb);
        bf16x16 bk3 = load_b_frag(K1 + 32 + kb);
        f32x8 s0 = f32x8(0.f), s1 = f32x8(0.f);
        s0 = wmma_bf16(aq0, bk0, s0);
        s0 = wmma_bf16(aq1, bk1, s0);
        s1 = wmma_bf16(aq0, bk2, s1);
        s1 = wmma_bf16(aq1, bk3, s1);

        // ---- online softmax over this 32-key chunk (rows live in vgpr index + lane half)
        float alpha[8];
#pragma unroll
        for (int r = 0; r < 8; ++r) {
            float v0 = s0[r] * INV_SQRT_DK;
            float v1 = s1[r] * INV_SQRT_DK;
            float cm = fmaxf(v0, v1);
            cm = fmaxf(cm, __shfl_xor(cm, 1));
            cm = fmaxf(cm, __shfl_xor(cm, 2));
            cm = fmaxf(cm, __shfl_xor(cm, 4));
            cm = fmaxf(cm, __shfl_xor(cm, 8));
            float mn = fmaxf(m[r], cm);
            alpha[r] = __expf(m[r] - mn);
            m[r] = mn;
            float p0 = __expf(v0 - mn);
            float p1 = __expf(v1 - mn);
            s0[r] = p0; s1[r] = p1;
            float rs = p0 + p1;
            rs += __shfl_xor(rs, 1);
            rs += __shfl_xor(rs, 2);
            rs += __shfl_xor(rs, 4);
            rs += __shfl_xor(rs, 8);
            l[r] = l[r] * alpha[r] + rs;
        }
#pragma unroll
        for (int f = 0; f < 4; ++f)
#pragma unroll
            for (int r = 0; r < 8; ++r) acc[f][r] *= alpha[r];

        // ---- stage P (C-layout) into LDS, reload as A-fragment
#pragma unroll
        for (int r = 0; r < 8; ++r) {
            const int row = r + (hi ? 8 : 0);
            P[row * 32 + l16]      = f2bf(s0[r]);
            P[row * 32 + 16 + l16] = f2bf(s1[r]);
        }
        asm volatile("s_wait_dscnt 0x0" ::: "memory");
        bf16x16 pf = load_a_frag(P + l16 * 32 + ak);

        // ---- ctx += P * V  (V pre-transposed so B fragments are contiguous)
        bf16x16 bv0 = load_b_frag(Vb + (size_t)(0 * 16 + l16) * SEQ + k0 + kb);
        bf16x16 bv1 = load_b_frag(Vb + (size_t)(1 * 16 + l16) * SEQ + k0 + kb);
        bf16x16 bv2 = load_b_frag(Vb + (size_t)(2 * 16 + l16) * SEQ + k0 + kb);
        bf16x16 bv3 = load_b_frag(Vb + (size_t)(3 * 16 + l16) * SEQ + k0 + kb);
        acc[0] = wmma_bf16(pf, bv0, acc[0]);
        acc[1] = wmma_bf16(pf, bv1, acc[1]);
        acc[2] = wmma_bf16(pf, bv2, acc[2]);
        acc[3] = wmma_bf16(pf, bv3, acc[3]);
    }

    // ---- normalize and store ctx row-major [token][h*64+d]
    const int b = bh >> 4, h = bh & (NUM_HEADS - 1);
    float inv[8];
#pragma unroll
    for (int r = 0; r < 8; ++r) inv[r] = 1.0f / l[r];
#pragma unroll
    for (int f = 0; f < 4; ++f)
#pragma unroll
        for (int r = 0; r < 8; ++r) {
            const int row = b * SEQ + qbase + r + (hi ? 8 : 0);
            const int col = h * D_K + f * 16 + l16;
            ctx[(size_t)row * D_MODEL + col] = f2bf(acc[f][r] * inv[r]);
        }
}

// ---------- host side ----------
static inline size_t align256(size_t x) { return (x + 255) & ~(size_t)255; }

extern "C" void kernel_launch(void* const* d_in, const int* in_sizes, int n_in,
                              void* d_out, int out_size, void* d_ws, size_t ws_size,
                              hipStream_t stream) {
    const float* X  = (const float*)d_in[0];
    const float* Y  = (const float*)d_in[1];
    const float* Wq = (const float*)d_in[2];  const float* bq = (const float*)d_in[3];
    const float* Wk = (const float*)d_in[4];  const float* bk = (const float*)d_in[5];
    const float* Wv = (const float*)d_in[6];  const float* bv = (const float*)d_in[7];
    const float* Wg = (const float*)d_in[8];  const float* bg = (const float*)d_in[9];
    const float* Wb = (const float*)d_in[10]; const float* bb = (const float*)d_in[11];
    const float* Wo = (const float*)d_in[12]; const float* bo = (const float*)d_in[13];

    float* Xnew = (float*)d_out;                           // 8192*1024
    float* Ynew = (float*)d_out + (size_t)M_TOK * D_MODEL; // 8192*1024

    // workspace carve-up
    char* base = (char*)d_ws;
    size_t off = 0;
    auto carve = [&](size_t bytes) { char* p = base + off; off = align256(off + bytes); return p; };
    const size_t wbytes = (size_t)D_MODEL * D_MODEL * 2;
    const size_t abytes = (size_t)M_TOK * D_MODEL * 2;
    __bf16* wqt  = (__bf16*)carve(wbytes);
    __bf16* wkt  = (__bf16*)carve(wbytes);
    __bf16* wvt  = (__bf16*)carve(wbytes);
    __bf16* wgbt = (__bf16*)carve(wbytes);
    __bf16* wot  = (__bf16*)carve(wbytes);
    float*  bgb  = (float*)carve(D_MODEL * 4);
    __bf16* xbf  = (__bf16*)carve(abytes);
    __bf16* ybf  = (__bf16*)carve(abytes);
    __bf16* qb   = (__bf16*)carve(abytes);
    __bf16* kbuf = (__bf16*)carve(abytes);
    __bf16* vtb  = (__bf16*)carve(abytes);
    __bf16* ctxb = (__bf16*)carve(abytes);
    __bf16* tb   = (__bf16*)carve(abytes);
    __bf16* ynbf = (__bf16*)carve(abytes);
    (void)ws_size; (void)in_sizes; (void)n_in; (void)out_size;

    // prep
    prep_weights<<<(D_MODEL * D_MODEL) / 256, 256, 0, stream>>>(
        Wq, Wk, Wv, Wg, Wb, Wo, bg, bb, wqt, wkt, wvt, wgbt, wot, bgb);
    cvt_f32_bf16<<<(M_TOK * D_MODEL / 4) / 256, 256, 0, stream>>>(X, xbf);
    cvt_f32_bf16<<<(M_TOK * D_MODEL / 4) / 256, 256, 0, stream>>>(Y, ybf);

    const dim3 ggrid(M_TOK / 128, D_MODEL / 64);

    // ---- pass 1: queries from X, keys/values from Y -> Y_new
    gemm_bf16<<<ggrid, 256, 0, stream>>>(xbf, wqt, bq, qb,   nullptr, 1);
    gemm_bf16<<<ggrid, 256, 0, stream>>>(ybf, wkt, bk, kbuf, nullptr, 1);
    gemm_bf16<<<ggrid, 256, 0, stream>>>(ybf, wvt, bv, vtb,  nullptr, 2);
    attn_flash<<<M_TOK / 16 / 8, 256, 0, stream>>>(qb, kbuf, vtb, ctxb);
    gemm_bf16<<<ggrid, 256, 0, stream>>>(ctxb, wgbt, bgb, tb, nullptr, 0);
    gemm_bf16<<<ggrid, 256, 0, stream>>>(tb,   wot,  bo,  ynbf, Ynew, 4);

    // ---- pass 2: queries from Y_new, keys/values from X -> X_new
    gemm_bf16<<<ggrid, 256, 0, stream>>>(ynbf, wqt, bq, qb,   nullptr, 1);
    gemm_bf16<<<ggrid, 256, 0, stream>>>(xbf,  wkt, bk, kbuf, nullptr, 1);
    gemm_bf16<<<ggrid, 256, 0, stream>>>(xbf,  wvt, bv, vtb,  nullptr, 2);
    attn_flash<<<M_TOK / 16 / 8, 256, 0, stream>>>(qb, kbuf, vtb, ctxb);
    gemm_bf16<<<ggrid, 256, 0, stream>>>(ctxb, wgbt, bgb, tb, nullptr, 0);
    gemm_bf16<<<ggrid, 256, 0, stream>>>(tb,   wot,  bo,  nullptr, Xnew, 3);
}
